// NeuralCDE_31069793419749
// MI455X (gfx1250) — compile-verified, matrix operand order checked by reading, not run on hardware
//
#include <hip/hip_runtime.h>
#include <math.h>

typedef __attribute__((ext_vector_type(16))) _Float16 v16h;
typedef __attribute__((ext_vector_type(8)))  _Float16 v8h;
typedef __attribute__((ext_vector_type(8)))  float    v8f;

namespace {
constexpr int B_  = 2048;
constexpr int T_  = 100;
constexpr int C_  = 8;
constexpr int H_  = 64;      // z width
constexpr int HH_ = 128;     // hidden width
constexpr int FO_ = H_ * C_; // 512, W_out fan-out
constexpr int O_  = 10;
constexpr int NSTEP = T_ - 1;

// ---- LDS layout (bytes); every offset 16B aligned ----
constexpr int SM_WT_IN  = 0;                              // [128 n][64 k] f16
constexpr int SM_WT_H0  = SM_WT_IN  + HH_ * H_  * 2;      // [128][128] f16
constexpr int SM_WT_H1  = SM_WT_H0  + HH_ * HH_ * 2;      // [128][128] f16
constexpr int SM_WT_OUT = SM_WT_H1  + HH_ * HH_ * 2;      // [512 n][128 k] f16
constexpr int SM_ZF     = SM_WT_OUT + FO_ * HH_ * 2;      // [16][64] f32  current z
constexpr int SM_KACC   = SM_ZF     + 16 * H_ * 4;        // [16][64] f32  RK4 accum
constexpr int SM_GK     = SM_KACC   + 16 * H_ * 4;        // [16][64] f32  stage output
constexpr int SM_ZE     = SM_GK     + 16 * H_ * 4;        // [16][64] f16  z_eval (A op)
constexpr int SM_HA     = SM_ZE     + 16 * H_ * 2;        // [16][128] f16
constexpr int SM_HB     = SM_HA     + 16 * HH_ * 2;       // [16][128] f16
constexpr int SM_DXDT   = SM_HB     + 16 * HH_ * 2;       // [16][8] f32
constexpr int SM_LOGITS = SM_DXDT   + 16 * C_ * 4;        // [16][10] f32
constexpr int SM_BYTES  = SM_LOGITS + 16 * O_ * 4;        // ~231 KB < 320 KB/WGP
} // namespace

// Load one 16x32 f16 fragment (A or B operand) from a row-major [rows][Kdim]
// f16 array. CDNA5 16-bit 16x32 layout: lane = r(0..15) + 16*khalf;
// v0..v3 hold K = khalf*8 + 0..7, v4..v7 hold K = 16 + khalf*8 + 0..7.
// Two contiguous 16-byte ds loads per lane.
__device__ __forceinline__ v16h ld_frag(const _Float16* base, int Kdim,
                                        int row_tile, int kc, int lane) {
  const _Float16* p =
      base + (row_tile * 16 + (lane & 15)) * Kdim + kc * 32 + (lane >> 4) * 8;
  v8h lo = *(const v8h*)p;
  v8h hi = *(const v8h*)(p + 16);
  return __builtin_shufflevector(lo, hi, 0, 1, 2, 3, 4, 5, 6, 7, 8, 9, 10, 11,
                                 12, 13, 14, 15);
}

__global__ void zero_out_kernel(float* out) {
  if (threadIdx.x < 2) out[threadIdx.x] = 0.0f;
}

__global__ void __launch_bounds__(256, 1)
neural_cde_kernel(const float* __restrict__ coeffs, const int* __restrict__ y,
                  const float* __restrict__ times,
                  const float* __restrict__ W_init, const float* __restrict__ b_init,
                  const float* __restrict__ W_in,   const float* __restrict__ b_in,
                  const float* __restrict__ W_h,    const float* __restrict__ b_h,
                  const float* __restrict__ W_out,  const float* __restrict__ b_out,
                  const float* __restrict__ W_read, const float* __restrict__ b_read,
                  float* __restrict__ out) {
  extern __shared__ char smem[];
  _Float16* wt_in  = (_Float16*)(smem + SM_WT_IN);
  _Float16* wt_h0  = (_Float16*)(smem + SM_WT_H0);
  _Float16* wt_h1  = (_Float16*)(smem + SM_WT_H1);
  _Float16* wt_out = (_Float16*)(smem + SM_WT_OUT);
  float* zf_s   = (float*)(smem + SM_ZF);
  float* kacc_s = (float*)(smem + SM_KACC);
  float* gk_s   = (float*)(smem + SM_GK);
  _Float16* ze_s = (_Float16*)(smem + SM_ZE);
  _Float16* hA_s = (_Float16*)(smem + SM_HA);
  _Float16* hB_s = (_Float16*)(smem + SM_HB);
  float* dx_s   = (float*)(smem + SM_DXDT);
  float* lg_s   = (float*)(smem + SM_LOGITS);

  const int tid  = threadIdx.x;
  const int lane = tid & 31;
  const int wave = tid >> 5;             // 0..7
  const int nl   = lane & 15;
  const int hi   = lane >> 4;
  const int bm0  = blockIdx.x * 16;      // this block's 16 batch rows

  // ---- one-time: weights -> LDS, f16, n-major (B-operand friendly) ----
  for (int idx = tid; idx < H_ * HH_; idx += 256) {     // W_in [64][128]
    int k = idx >> 7, n = idx & 127;
    wt_in[n * H_ + k] = (_Float16)W_in[idx];
  }
  for (int idx = tid; idx < HH_ * HH_; idx += 256) {    // W_h[0], W_h[1]
    int k = idx >> 7, n = idx & 127;
    wt_h0[n * HH_ + k] = (_Float16)W_h[idx];
    wt_h1[n * HH_ + k] = (_Float16)W_h[HH_ * HH_ + idx];
  }
  for (int idx = tid; idx < HH_ * FO_; idx += 256) {    // W_out [128][512]
    int k = idx >> 9, n = idx & 511;
    wt_out[n * HH_ + k] = (_Float16)W_out[idx];
  }

  // ---- z0 = coeffs[:,0] @ W_init + b_init ----
  for (int idx = tid; idx < 16 * H_; idx += 256) {
    int m = idx >> 6, h = idx & 63;
    int b = bm0 + m;
    float s = b_init[h];
#pragma unroll
    for (int c = 0; c < C_; ++c)
      s += coeffs[(b * T_) * C_ + c] * W_init[c * H_ + h];
    zf_s[idx] = s;
  }

  // ---- loop-invariant per-lane registers: biases + B fragments ----
  const float bv_in = b_in[wave * 16 + nl];
  const float bv_h0 = b_h[wave * 16 + nl];
  const float bv_h1 = b_h[HH_ + wave * 16 + nl];
  float bv_out[4];
#pragma unroll
  for (int tt = 0; tt < 4; ++tt)
    bv_out[tt] = b_out[(wave * 4 + tt) * 16 + nl];

  __syncthreads();

  // Register-resident B operands for the three hidden layers (80 VGPRs):
  // they are reused 396 times, and LDS limits us to 1 WG/WGP anyway.
  v16h wfin[2], wfh0[4], wfh1[4];
#pragma unroll
  for (int c = 0; c < H_ / 32; ++c)
    wfin[c] = ld_frag(wt_in, H_, wave, c, lane);
#pragma unroll
  for (int c = 0; c < HH_ / 32; ++c) {
    wfh0[c] = ld_frag(wt_h0, HH_, wave, c, lane);
    wfh1[c] = ld_frag(wt_h1, HH_, wave, c, lane);
  }

  const float alpha_next[4] = {0.5f, 0.5f, 1.0f, 0.0f};
  const float wgt[4]        = {1.0f, 2.0f, 2.0f, 1.0f};

  // ================= time loop =================
#pragma unroll 1
  for (int i = 0; i < NSTEP; ++i) {
    const float dt = times[i + 1] - times[i];
    const float inv_dt = 1.0f / dt;
    // dX/dt for this interval and z_eval for stage 0
    if (tid < 16 * C_) {
      int m = tid >> 3, c = tid & 7;
      int b = bm0 + m;
      dx_s[tid] = (coeffs[(b * T_ + i + 1) * C_ + c] -
                   coeffs[(b * T_ + i) * C_ + c]) * inv_dt;
    }
    for (int idx = tid; idx < 16 * H_; idx += 256)
      ze_s[idx] = (_Float16)zf_s[idx];
    __syncthreads();

    // dxdt values this lane needs for the f-stage: dx[r + 8*hi][lane&7]
    float dreg[8];
#pragma unroll
    for (int r = 0; r < 8; ++r) dreg[r] = dx_s[((r + 8 * hi) << 3) + (lane & 7)];

    // -------- RK4 stages --------
#pragma unroll 1
    for (int s = 0; s < 4; ++s) {
      // h1 = relu(z_eval @ W_in + b_in): wave owns N-tile `wave`
      {
        v8f acc;
#pragma unroll
        for (int r = 0; r < 8; ++r) acc[r] = bv_in;
#pragma unroll
        for (int c = 0; c < H_ / 32; ++c) {
          v16h a = ld_frag(ze_s, H_, 0, c, lane);
          acc = __builtin_amdgcn_wmma_f32_16x16x32_f16(false, a, false, wfin[c],
                                                       (short)0, acc, false, false);
        }
#pragma unroll
        for (int r = 0; r < 8; ++r)
          hA_s[(r + 8 * hi) * HH_ + wave * 16 + nl] = (_Float16)fmaxf(acc[r], 0.0f);
      }
      __syncthreads();
      // h2 = relu(h1 @ W_h0 + b_h0)
      {
        v8f acc;
#pragma unroll
        for (int r = 0; r < 8; ++r) acc[r] = bv_h0;
#pragma unroll
        for (int c = 0; c < HH_ / 32; ++c) {
          v16h a = ld_frag(hA_s, HH_, 0, c, lane);
          acc = __builtin_amdgcn_wmma_f32_16x16x32_f16(false, a, false, wfh0[c],
                                                       (short)0, acc, false, false);
        }
#pragma unroll
        for (int r = 0; r < 8; ++r)
          hB_s[(r + 8 * hi) * HH_ + wave * 16 + nl] = (_Float16)fmaxf(acc[r], 0.0f);
      }
      __syncthreads();
      // h3 = relu(h2 @ W_h1 + b_h1)
      {
        v8f acc;
#pragma unroll
        for (int r = 0; r < 8; ++r) acc[r] = bv_h1;
#pragma unroll
        for (int c = 0; c < HH_ / 32; ++c) {
          v16h a = ld_frag(hB_s, HH_, 0, c, lane);
          acc = __builtin_amdgcn_wmma_f32_16x16x32_f16(false, a, false, wfh1[c],
                                                       (short)0, acc, false, false);
        }
#pragma unroll
        for (int r = 0; r < 8; ++r)
          hA_s[(r + 8 * hi) * HH_ + wave * 16 + nl] = (_Float16)fmaxf(acc[r], 0.0f);
      }
      __syncthreads();
      // f = tanh(h3 @ W_out + b_out); g = einsum('bhc,bc->bh', f, dxdt)
      // wave owns columns [wave*64, wave*64+64) -> h in [wave*8, wave*8+8)
      {
        v16h af[4];                      // A fragments shared by all 4 tiles
#pragma unroll
        for (int c = 0; c < HH_ / 32; ++c)
          af[c] = ld_frag(hA_s, HH_, 0, c, lane);
#pragma unroll
        for (int tt = 0; tt < 4; ++tt) {
          int ntile = wave * 4 + tt;
          v8f acc;
#pragma unroll
          for (int r = 0; r < 8; ++r) acc[r] = bv_out[tt];
#pragma unroll
          for (int c = 0; c < HH_ / 32; ++c) {
            v16h b = ld_frag(wt_out, HH_, ntile, c, lane);
            acc = __builtin_amdgcn_wmma_f32_16x16x32_f16(false, af[c], false, b,
                                                         (short)0, acc, false, false);
          }
          int hglob = (ntile * 16 + nl) >> 3;   // n -> h (c = n & 7 = lane&7)
#pragma unroll
          for (int r = 0; r < 8; ++r) {
            float t = tanhf(acc[r]) * dreg[r];
            t += __shfl_xor(t, 1, 32);          // reduce over c within 8 lanes
            t += __shfl_xor(t, 2, 32);
            t += __shfl_xor(t, 4, 32);
            if ((lane & 7) == 0) gk_s[(r + 8 * hi) * H_ + hglob] = t;
          }
        }
      }
      __syncthreads();
      // RK4 bookkeeping: kacc += wgt[s]*g ; z_eval for next stage
      for (int idx = tid; idx < 16 * H_; idx += 256) {
        float gv = gk_s[idx];
        float ka = (s == 0) ? gv : kacc_s[idx] + wgt[s] * gv;
        kacc_s[idx] = ka;
        if (s < 3)
          ze_s[idx] = (_Float16)(zf_s[idx] + alpha_next[s] * dt * gv);
      }
      __syncthreads();
    }
    // z += dt/6 * (k1 + 2k2 + 2k3 + k4)
    const float dt6 = dt * (1.0f / 6.0f);
    for (int idx = tid; idx < 16 * H_; idx += 256)
      zf_s[idx] += dt6 * kacc_s[idx];
    __syncthreads();
  }

  // ================= readout: logits, log-softmax loss, accuracy =========
  if (tid < 16 * O_) {
    int m = tid / O_, o = tid % O_;
    float s = b_read[o];
#pragma unroll
    for (int h = 0; h < H_; ++h) s += zf_s[m * H_ + h] * W_read[h * O_ + o];
    lg_s[m * O_ + o] = s;
  }
  __syncthreads();
  if (tid < 16) {
    int m = tid;
    int yy = y[bm0 + m];
    float mx = lg_s[m * O_];
    int am = 0;
#pragma unroll
    for (int o = 1; o < O_; ++o) {
      float v = lg_s[m * O_ + o];
      if (v > mx) { mx = v; am = o; }
    }
    float lse = 0.0f;
#pragma unroll
    for (int o = 0; o < O_; ++o) lse += expf(lg_s[m * O_ + o] - mx);
    lse = logf(lse) + mx;
    float nll = lse - lg_s[m * O_ + yy];
    atomicAdd(&out[0], nll * (1.0f / (float)B_));
    atomicAdd(&out[1], (am == yy) ? 1.0f : 0.0f);
  }
}

extern "C" void kernel_launch(void* const* d_in, const int* in_sizes, int n_in,
                              void* d_out, int out_size, void* d_ws, size_t ws_size,
                              hipStream_t stream) {
  const float* coeffs = (const float*)d_in[0];
  const int*   yv     = (const int*)d_in[1];
  const float* times  = (const float*)d_in[2];
  const float* W_init = (const float*)d_in[3];
  const float* b_init = (const float*)d_in[4];
  const float* W_in   = (const float*)d_in[5];
  const float* b_in   = (const float*)d_in[6];
  const float* W_h    = (const float*)d_in[7];
  const float* b_h    = (const float*)d_in[8];
  const float* W_out  = (const float*)d_in[9];
  const float* b_out  = (const float*)d_in[10];
  const float* W_read = (const float*)d_in[11];
  const float* b_read = (const float*)d_in[12];
  float* out = (float*)d_out;

  zero_out_kernel<<<1, 32, 0, stream>>>(out);
  neural_cde_kernel<<<dim3(B_ / 16), dim3(256), SM_BYTES, stream>>>(
      coeffs, yv, times, W_init, b_init, W_in, b_in, W_h, b_h, W_out, b_out,
      W_read, b_read, out);
}
